// Model_28278064677428
// MI455X (gfx1250) — matrix operand order repaired and checked by reading, not compiled, not run on hardware
//
#include <hip/hip_runtime.h>
#include <math.h>

// ---------------- model constants ----------------
#define BB   32
#define LL0  512
#define CC   32
#define PP   96
#define SEG  16
#define DG   32      // D_GIN
#define DE   64      // D_ENC
#define DI   128     // D_INNER
#define DSN  16      // D_STATE
#define DTR  8       // DT_RANK
#define PVHN 40
#define KSUB 20
#define COLS (BB*SEG*DG)   // 16384

typedef __attribute__((ext_vector_type(16))) __bf16 v16bf;
typedef __attribute__((ext_vector_type(8)))  __bf16 v8bf;
typedef __attribute__((ext_vector_type(8)))  float  v8f;

__device__ __forceinline__ float bf2f(unsigned short h){
  return __uint_as_float(((unsigned int)h) << 16);
}
__device__ __forceinline__ unsigned short f2bf(float f){
  unsigned int u = __float_as_uint(f);
  u += 0x7fffu + ((u >> 16) & 1u);
  return (unsigned short)(u >> 16);
}
// 16-bf16 fragment from two 16B (b128) loads of 8 contiguous bf16 each
__device__ __forceinline__ v16bf ld_frag(const unsigned short* p0, const unsigned short* p1){
  v8bf lo = __builtin_bit_cast(v8bf, *(const uint4*)p0);
  v8bf hi = __builtin_bit_cast(v8bf, *(const uint4*)p1);
  return __builtin_shufflevector(lo, hi, 0,1,2,3,4,5,6,7,8,9,10,11,12,13,14,15);
}

// ---------------- generic WMMA GEMM (no bounds checks by construction) ----------------
// C[M,N] = A[M,K] * B[K,N], A row-major bf16 (lda mult of 8, K mult of 32),
// B given TRANSPOSED: BT[N,K] row-major bf16 (ldbt mult of 8).
// M multiple of 64; N an exact multiple of 16*NS*gridDim.x.
// 128 threads/block = 4 waves; wave w owns m-tile blockIdx.y*64 + w*16, NS n-subtiles.
template<int NS>
__global__ void k_gemm(const unsigned short* __restrict__ A, int lda,
                       const unsigned short* __restrict__ BT, int ldbt,
                       float* __restrict__ Cf, unsigned short* __restrict__ Cb, int ldc,
                       int K,
                       const float* __restrict__ bias,
                       const unsigned short* __restrict__ addbf,
                       const float* __restrict__ scale_ptr,
                       const float* __restrict__ addf,
                       int relu, int accum)
{
  const int lane  = threadIdx.x & 31;
  const int wave  = threadIdx.x >> 5;
  const int m0    = (blockIdx.y << 6) + (wave << 4);
  const int n0    = blockIdx.x * (NS << 4);
  const int mrow  = m0 + (lane & 15);
  const int a_klo = (lane & 16) ? 8  : 0;   // ISA 16-bit A 16x32 layout
  const int b_klo = (lane & 16) ? 8  : 0;   // ISA 16-bit B 32x16 layout (same k order as A, via BT)
  const int nlane = lane & 15;

  v8f acc[NS] = {};
  const unsigned short* ap = A + (long)mrow * lda + a_klo;

  for (int k0 = 0; k0 < K; k0 += 32) {
    v16bf af = ld_frag(ap + k0, ap + k0 + 16);
#pragma unroll
    for (int s = 0; s < NS; ++s) {
      const unsigned short* bp = BT + (long)(n0 + (s << 4) + nlane) * ldbt + k0 + b_klo;
      v16bf bf = ld_frag(bp, bp + 16);
      acc[s] = __builtin_amdgcn_wmma_f32_16x16x32_bf16(
                 false, af, false, bf, (short)0, acc[s], false, false);
    }
  }

  float sval = scale_ptr ? scale_ptr[0] : 1.0f;
#pragma unroll
  for (int s = 0; s < NS; ++s) {
    int nc = n0 + (s << 4) + nlane;
    float bv = bias ? bias[nc] : 0.0f;
#pragma unroll
    for (int r = 0; r < 8; ++r) {
      int mr = m0 + r + ((lane >> 4) << 3);   // ISA f32 C/D layout
      long idx = (long)mr * ldc + nc;
      float v = acc[s][r] + bv;
      if (addbf) v += sval * bf2f(addbf[idx]);
      if (addf)  v += addf[idx];
      if (relu)  v = fmaxf(v, 0.0f);
      if (Cf) Cf[idx] = accum ? (Cf[idx] + v) : v;
      if (Cb) Cb[idx] = f2bf(v);
    }
  }
}

// ---------------- elementwise / support kernels ----------------
__global__ void k_f2b(unsigned short* dst, const float* src, long n){
  long i = (long)blockIdx.x * blockDim.x + threadIdx.x;
  long st = (long)gridDim.x * blockDim.x;
  for (; i < n; i += st) dst[i] = f2bf(src[i]);
}
// convert + transpose + pad: dst[n*Kpad + k] = bf16(src[k*N + n]) (0 outside K x N)
__global__ void k_cvtT(unsigned short* __restrict__ dst, const float* __restrict__ src,
                       int K, int N, int Kpad, long total){
  long i = (long)blockIdx.x * blockDim.x + threadIdx.x;
  long st = (long)gridDim.x * blockDim.x;
  for (; i < total; i += st){
    int k = (int)(i % Kpad);
    int n = (int)(i / Kpad);
    unsigned short v = 0;
    if (k < K && n < N) v = f2bf(src[(long)k * N + n]);
    dst[i] = v;
  }
}
__global__ void k_zero_us(unsigned short* dst, long n){
  long i = (long)blockIdx.x * blockDim.x + threadIdx.x;
  long st = (long)gridDim.x * blockDim.x;
  for (; i < n; i += st) dst[i] = 0;
}
__global__ void k_prep_eps(const float* e0, const float* e1, float* out){
  if (threadIdx.x == 0){ out[0] = 1.f + e0[0]; out[1] = 1.f + e1[0]; }
}
__global__ void k_decomp(const float* __restrict__ x, float* __restrict__ res,
                         float* __restrict__ trend, long total){
  long i = (long)blockIdx.x * blockDim.x + threadIdx.x;
  long st = (long)gridDim.x * blockDim.x;
  for (; i < total; i += st){
    int c = (int)(i % CC);
    int l = (int)((i / CC) % LL0);
    int b = (int)(i / ((long)CC * LL0));
    float s = 0.f;
    for (int j = -12; j <= 12; ++j){
      int ll = l + j; ll = ll < 0 ? 0 : (ll >= LL0 ? LL0 - 1 : ll);
      s += x[((long)b * LL0 + ll) * CC + c];
    }
    float m = s * (1.0f / 25.0f);
    trend[i] = m; res[i] = x[i] - m;
  }
}
__global__ void k_down(const float* __restrict__ src, float* __restrict__ dst,
                       int Lout, long total){
  long i = (long)blockIdx.x * blockDim.x + threadIdx.x;
  long st = (long)gridDim.x * blockDim.x;
  for (; i < total; i += st){
    int c = (int)(i % CC);
    long r = i / CC;
    int l = (int)(r % Lout);
    int b = (int)(r / Lout);
    long s0 = ((long)b * (2 * Lout) + 2 * l) * CC + c;
    dst[i] = 0.5f * (src[s0] + src[s0 + CC]);
  }
}
__global__ void k_trendT(const float* __restrict__ trend, unsigned short* __restrict__ dstb,
                         long total){
  long i = (long)blockIdx.x * blockDim.x + threadIdx.x;
  long st = (long)gridDim.x * blockDim.x;
  for (; i < total; i += st){
    int l = (int)(i % LL0);
    long r = i / LL0;
    int c = (int)(r % CC);
    int b = (int)(r / CC);
    dstb[i] = f2bf(trend[((long)b * LL0 + l) * CC + c]);
  }
}
// node embeddings: n = tanh(3*(e@l)), written padded [N,64] (cols 40..63 = 0)
__global__ void k_n12(const float* __restrict__ e, const float* __restrict__ lm,
                      unsigned short* __restrict__ nb, int N){
  int i = blockIdx.x * blockDim.x + threadIdx.x;
  if (i >= N * 64) return;
  int row = i / 64, col = i % 64;
  unsigned short out = 0;
  if (col < PVHN){
    float s = 0.f;
    for (int k = 0; k < PVHN; ++k) s += e[row * PVHN + k] * lm[k * PVHN + col];
    out = f2bf(tanhf(3.0f * s));
  }
  nb[i] = out;
}
__global__ void k_adj(const float* __restrict__ C1, float* __restrict__ adj,
                      float* __restrict__ adjw, int N){
  long total = (long)N * N;
  long i = (long)blockIdx.x * blockDim.x + threadIdx.x;
  long st = (long)gridDim.x * blockDim.x;
  for (; i < total; i += st){
    int row = (int)(i / N), col = (int)(i % N);
    float a = C1[(long)row * N + col] - C1[(long)col * N + row]; // n2 n1^T = (n1 n2^T)^T
    float v = fmaxf(tanhf(3.0f * a), 0.0f);
    adj[i] = v; adjw[i] = v;
  }
}
__global__ void k_topk(const float* __restrict__ adj, float* __restrict__ adjw,
                       unsigned short* __restrict__ adjb, int N){
  int row = blockIdx.x;
  int tid = threadIdx.x;
  __shared__ float sv[256];
  __shared__ int   si[256];
  for (int j = tid; j < N; j += 256) adjb[(long)row * N + j] = 0;
  __syncthreads();
  for (int it = 0; it < KSUB; ++it){
    float best = -1e30f; int bi = -1;
    for (int j = tid; j < N; j += 256){
      float v = adjw[(long)row * N + j];
      if (v > best){ best = v; bi = j; }
    }
    sv[tid] = best; si[tid] = bi;
    __syncthreads();
    for (int off = 128; off > 0; off >>= 1){
      if (tid < off && sv[tid + off] > sv[tid]){ sv[tid] = sv[tid + off]; si[tid] = si[tid + off]; }
      __syncthreads();
    }
    if (tid == 0 && si[0] >= 0){
      int j = si[0];
      adjb[(long)row * N + j] = f2bf(adj[(long)row * N + j]);
      adjw[(long)row * N + j] = -1e30f;
    }
    __syncthreads();
  }
}
// H0 in both layouts: normal [n,(b,s,d)] and transposed HT0[(b,s,d), n]
__global__ void k_h0(const float* __restrict__ xs, const float* __restrict__ sw,
                     const float* __restrict__ sb, unsigned short* __restrict__ H0,
                     unsigned short* __restrict__ HT0, int L, int N, long total){
  long i = (long)blockIdx.x * blockDim.x + threadIdx.x;
  long st = (long)gridDim.x * blockDim.x;
  for (; i < total; i += st){
    int d = (int)(i % DG);
    long r = i / DG;
    int s = (int)(r % SEG); r /= SEG;
    int b = (int)(r % BB);
    int n = (int)(r / BB);
    int f = s * N + n;
    int l = f / CC, c = f % CC;
    unsigned short v = f2bf(xs[((long)b * L + l) * CC + c] * sw[d] + sb[d]);
    H0[i] = v;
    int col = (b * SEG + s) * DG + d;
    HT0[(long)col * N + n] = v;
  }
}
// transpose activation: HT[(b,s,d), n] = H[n,(b,s,d)]
__global__ void k_transposeH(const unsigned short* __restrict__ H,
                             unsigned short* __restrict__ HT, int N, long total){
  long i = (long)blockIdx.x * blockDim.x + threadIdx.x;
  long st = (long)gridDim.x * blockDim.x;
  for (; i < total; i += st){
    int m  = (int)(i % N);
    long col = i / N;
    long bs = col / DG;
    int d  = (int)(col % DG);
    HT[i] = H[((long)m * (BB * SEG) + bs) * DG + d];
  }
}
// xr -> padded [row,32] A-operand for the embedding GEMM (cols 16..31 pre-zeroed)
__global__ void k_xr(const unsigned short* __restrict__ H, const float* __restrict__ xs,
                     const float* __restrict__ sw, const float* __restrict__ sb,
                     const float* __restrict__ rew, const float* __restrict__ reb,
                     unsigned short* __restrict__ xrb, int L, int N, int Lp, long total){
  long i = (long)blockIdx.x * blockDim.x + threadIdx.x;
  long st = (long)gridDim.x * blockDim.x;
  for (; i < total; i += st){
    int l = (int)(i % L);
    long r = i / L;
    int c = (int)(r % CC);
    int b = (int)(r / CC);
    int f = l * CC + c;
    int s = f / N, n = f % N;
    long base = (((long)n * BB + b) * SEG + s) * DG;
    float xo = xs[((long)b * L + l) * CC + c];
    float acc = reb[0];
    for (int d = 0; d < DG; ++d)
      acc += (bf2f(H[base + d]) + (xo * sw[d] + sb[d])) * rew[d];
    long row = ((long)b * CC + c) * Lp + (l >> 4);
    xrb[row * 32 + (l & 15)] = f2bf(acc);
  }
}
__global__ void k_rmsnorm(const float* __restrict__ u, const float* __restrict__ nw,
                          unsigned short* __restrict__ xb){
  long row = blockIdx.x;
  int tid = threadIdx.x;      // 0..63
  __shared__ float red[64];
  float v = u[row * DE + tid];
  red[tid] = v * v;
  __syncthreads();
  for (int off = 32; off > 0; off >>= 1){
    if (tid < off) red[tid] += red[tid + off];
    __syncthreads();
  }
  float r = rsqrtf(red[0] / (float)DE + 1e-6f);
  xb[row * DE + tid] = f2bf(v * r * nw[tid]);
}
__global__ void k_conv(const float* __restrict__ xz, const float* __restrict__ cw,
                       const float* __restrict__ cb, float* __restrict__ xc,
                       unsigned short* __restrict__ xcb, int Lp, long total){
  long i = (long)blockIdx.x * blockDim.x + threadIdx.x;
  long st = (long)gridDim.x * blockDim.x;
  for (; i < total; i += st){
    int d = (int)(i % DI);
    long row = i / DI;
    int t = (int)(row % Lp);
    long bq = row / Lp;
    float s = cb[d];
#pragma unroll
    for (int j = 0; j < 4; ++j){
      int tt = t + j - 3;
      if (tt >= 0) s += xz[((bq * Lp + tt) * (2 * DI)) + d] * cw[d * 4 + j];
    }
    float v = s / (1.f + expf(-s));  // silu
    xc[i] = v; xcb[i] = f2bf(v);
  }
}
__global__ void k_dtk(const float* __restrict__ dbc, const float* __restrict__ dtw,
                      const float* __restrict__ dtbias, float* __restrict__ dtout,
                      long total){
  long i = (long)blockIdx.x * blockDim.x + threadIdx.x;
  long st = (long)gridDim.x * blockDim.x;
  for (; i < total; i += st){
    int d = (int)(i % DI);
    long row = i / DI;
    float s = dtbias[d];
#pragma unroll
    for (int r = 0; r < DTR; ++r) s += dbc[row * 48 + r] * dtw[r * DI + d];
    dtout[i] = (s > 20.f) ? s : log1pf(expf(s));
  }
}
// selective scan: 1 block / sequence, thread d owns state row h[d, 0..15]
__global__ void k_scan(const float* __restrict__ dtb, const float* __restrict__ dbc,
                       const float* __restrict__ xc, const float* __restrict__ xz,
                       const float* __restrict__ Alog, const float* __restrict__ Dp,
                       unsigned short* __restrict__ gb, int Lp){
  int bq = blockIdx.x;
  int d  = threadIdx.x;    // 0..127
  float a[DSN], h[DSN];
#pragma unroll
  for (int n = 0; n < DSN; ++n){ a[n] = -expf(Alog[d * DSN + n]); h[n] = 0.f; }
  __shared__ float sBC[2 * DSN];
  for (int t = 0; t < Lp; ++t){
    long row = (long)bq * Lp + t;
    if (d < 2 * DSN) sBC[d] = dbc[row * 48 + DTR + d];
    __syncthreads();
    float dtv = dtb[row * DI + d];
    float xcv = xc[row * DI + d];
    float acc = 0.f;
#pragma unroll
    for (int n = 0; n < DSN; ++n){
      h[n] = expf(dtv * a[n]) * h[n] + dtv * sBC[n] * xcv;
      acc += h[n] * sBC[DSN + n];
    }
    float y  = acc + Dp[d] * xcv;
    float z  = xz[row * (2 * DI) + DI + d];
    float sz = z / (1.f + expf(-z));
    gb[row * DI + d] = f2bf(y * sz);
    __syncthreads();
  }
}
__global__ void k_final(const float* __restrict__ acc, const float* __restrict__ trendo,
                        float* __restrict__ out, long total){
  long i = (long)blockIdx.x * blockDim.x + threadIdx.x;
  long st = (long)gridDim.x * blockDim.x;
  for (; i < total; i += st){
    int c = (int)(i % CC);
    long r = i / CC;
    int p = (int)(r % PP);
    int b = (int)(r / PP);
    long s = ((long)b * CC + c) * PP + p;
    out[i] = acc[s] * (1.0f / 3.0f) + trendo[s];
  }
}

// ---------------- host orchestration ----------------
static inline unsigned gsb(long n){
  long b = (n + 255) / 256;
  if (b < 1) b = 1;
  if (b > 65535) b = 65535;
  return (unsigned)b;
}
static inline unsigned cdivu(long a, long b){ return (unsigned)((a + b - 1) / b); }

static void GEMM(hipStream_t st,
                 const unsigned short* A, int lda,
                 const unsigned short* BT, int ldbt,
                 float* Cf, unsigned short* Cb, int ldc,
                 long M, int N, int K,
                 const float* bias, const unsigned short* addbf, const float* scale_ptr,
                 const float* addf, int relu, int accum){
  // M % 64 == 0 guaranteed by construction; K % 32 == 0; N % 16 == 0.
  unsigned gy = (unsigned)(M / 64);
  if (N % 64 == 0){
    k_gemm<4><<<dim3((unsigned)(N / 64), gy), 128, 0, st>>>(A, lda, BT, ldbt, Cf, Cb, ldc, K,
                                                            bias, addbf, scale_ptr, addf, relu, accum);
  } else if (N % 32 == 0){
    k_gemm<2><<<dim3((unsigned)(N / 32), gy), 128, 0, st>>>(A, lda, BT, ldbt, Cf, Cb, ldc, K,
                                                            bias, addbf, scale_ptr, addf, relu, accum);
  } else {
    k_gemm<1><<<dim3((unsigned)(N / 16), gy), 128, 0, st>>>(A, lda, BT, ldbt, Cf, Cb, ldc, K,
                                                            bias, addbf, scale_ptr, addf, relu, accum);
  }
}

extern "C" void kernel_launch(void* const* d_in, const int* in_sizes, int n_in,
                              void* d_out, int out_size, void* d_ws, size_t ws_size,
                              hipStream_t stream){
  (void)in_sizes; (void)n_in; (void)out_size;
  auto F = [&](int i) -> const float* { return (const float*)d_in[i]; };

  const float* x       = F(0);
  const float* start_w = F(1);
  const float* start_b = F(2);
  const float* e1s[3] = {F(3),  F(7),  F(11)};
  const float* e2s[3] = {F(4),  F(8),  F(12)};
  const float* l1s[3] = {F(5),  F(9),  F(13)};
  const float* l2s[3] = {F(6),  F(10), F(14)};
  const float* eps0 = F(15); const float* W1_0 = F(16); const float* b1_0 = F(17);
  const float* W2_0 = F(18); const float* b2_0 = F(19);
  const float* eps1 = F(20); const float* W1_1 = F(21); const float* b1_1 = F(22);
  const float* W2_1 = F(23); const float* b2_1 = F(24);
  const float* re_w = F(25); const float* re_b = F(26);
  const float* pe_w = F(27); const float* pe_b = F(28);
  const float* rpe_wf = F(29); const float* rpe_b = F(30);
  const float* spe_w = F(31); const float* spe_b = F(32);
  const float* srpe_wf = F(33); const float* srpe_b = F(34);
  const float* mnorm[2] = {F(35), F(45)};
  const float* minp[2]  = {F(36), F(46)};
  const float* mcw[2]   = {F(37), F(47)};
  const float* mcb[2]   = {F(38), F(48)};
  const float* mxp[2]   = {F(39), F(49)};
  const float* mdtw[2]  = {F(40), F(50)};
  const float* mdtb[2]  = {F(41), F(51)};
  const float* mAlog[2] = {F(42), F(52)};
  const float* mD[2]    = {F(43), F(53)};
  const float* mop[2]   = {F(44), F(54)};
  const float* predw[3] = {F(55), F(57), F(59)};
  const float* predb[3] = {F(56), F(58), F(60)};
  const float* trend_w  = F(61); const float* trend_b = F(62);

  // ---- workspace carving ----
  char* base = (char*)d_ws;
  size_t off = 0;
  auto A8 = [&](size_t bytes) -> char* {
    char* p = base + off; off += (bytes + 255) & ~(size_t)255; return p;
  };
  const long Mmax = 1024L * 32;   // (B*C) * max(L/SEG)

  float* res    = (float*)A8((size_t)BB * LL0 * CC * 4);
  float* trend  = (float*)A8((size_t)BB * LL0 * CC * 4);
  float* xs1    = (float*)A8((size_t)BB * (LL0/2) * CC * 4);
  float* xs2    = (float*)A8((size_t)BB * (LL0/4) * CC * 4);
  unsigned short* trendTb = (unsigned short*)A8((size_t)BB * CC * LL0 * 2);
  float* trendo = (float*)A8((size_t)BB * CC * PP * 4);
  float* accp   = (float*)A8((size_t)BB * CC * PP * 4);
  unsigned short* xrb = (unsigned short*)A8((size_t)Mmax * 32 * 2);  // padded K=32
  float* eps1p  = (float*)A8(2 * 4);
  // transposed / padded bf16 weights (BT layout: [N, Kpad])
  unsigned short* twT   = (unsigned short*)A8((size_t)PP * LL0 * 2);       // [96,512]
  unsigned short* peT   = (unsigned short*)A8((size_t)DE * 32 * 2);        // [64,32]
  unsigned short* speT  = (unsigned short*)A8((size_t)DE * 32 * 2);
  unsigned short* rpeT  = (unsigned short*)A8((size_t)SEG * DE * 2);       // [16,64]
  unsigned short* srpeT = (unsigned short*)A8((size_t)SEG * DE * 2);
  unsigned short* ipT[2]  = {(unsigned short*)A8((size_t)(2*DI) * DE * 2), // [256,64]
                             (unsigned short*)A8((size_t)(2*DI) * DE * 2)};
  unsigned short* xpT[2]  = {(unsigned short*)A8((size_t)48 * DI * 2),     // [48,128]
                             (unsigned short*)A8((size_t)48 * DI * 2)};
  unsigned short* opT[2]  = {(unsigned short*)A8((size_t)DE * DI * 2),     // [64,128]
                             (unsigned short*)A8((size_t)DE * DI * 2)};
  unsigned short* w1T[2]  = {(unsigned short*)A8((size_t)DG * DG * 2),
                             (unsigned short*)A8((size_t)DG * DG * 2)};
  unsigned short* w2T[2]  = {(unsigned short*)A8((size_t)DG * DG * 2),
                             (unsigned short*)A8((size_t)DG * DG * 2)};
  unsigned short* predT[3] = {(unsigned short*)A8((size_t)PP * LL0 * 2),
                              (unsigned short*)A8((size_t)PP * (LL0/2) * 2),
                              (unsigned short*)A8((size_t)PP * (LL0/4) * 2)};

  size_t region0 = off;
  // --- GIN-phase region ---
  unsigned short* G0 = (unsigned short*)A8((size_t)1024 * COLS * 2);
  unsigned short* G1 = (unsigned short*)A8((size_t)1024 * COLS * 2);
  unsigned short* G2 = (unsigned short*)A8((size_t)1024 * COLS * 2);
  float* adjC = (float*)A8((size_t)1024 * 1024 * 4);
  float* adjM = (float*)A8((size_t)1024 * 1024 * 4);
  float* adjW = (float*)A8((size_t)1024 * 1024 * 4);
  unsigned short* adjB = (unsigned short*)A8((size_t)1024 * 1024 * 2);
  unsigned short* n1b  = (unsigned short*)A8((size_t)1024 * 64 * 2);   // padded K=64
  unsigned short* n2b  = (unsigned short*)A8((size_t)1024 * 64 * 2);
  size_t end_gin = off;
  // --- Mamba-phase region (aliases GIN region; disjoint lifetimes) ---
  off = region0;
  float* ubuf = (float*)A8((size_t)Mmax * DE * 4);
  unsigned short* xnb = (unsigned short*)A8((size_t)Mmax * DE * 2);
  float* xzb  = (float*)A8((size_t)Mmax * 2 * DI * 4);
  float* xcf  = (float*)A8((size_t)Mmax * DI * 4);
  unsigned short* xcb = (unsigned short*)A8((size_t)Mmax * DI * 2);
  float* dbc  = (float*)A8((size_t)Mmax * 48 * 4);     // padded N=48
  float* dtb  = (float*)A8((size_t)Mmax * DI * 4);
  unsigned short* gbb  = (unsigned short*)A8((size_t)Mmax * DI * 2);
  unsigned short* encb = (unsigned short*)A8((size_t)Mmax * DE * 2);
  unsigned short* ybf  = (unsigned short*)A8((size_t)BB * CC * LL0 * 2);
  size_t end_mm = off;
  size_t need = end_gin > end_mm ? end_gin : end_mm;
  if (need > ws_size) return;   // workspace too small: bail (no OOB)

  auto CVTT = [&](unsigned short* d, const float* s, int K, int N, int Kpad, int Npad){
    long tot = (long)Npad * Kpad;
    k_cvtT<<<gsb(tot), 256, 0, stream>>>(d, s, K, N, Kpad, tot);
  };

  // ---- weight conversions (transpose to BT layout) ----
  CVTT(twT, trend_w, LL0, PP, LL0, PP);
  CVTT(peT, pe_w, SEG, DE, 32, DE);     CVTT(speT, spe_w, SEG, DE, 32, DE);
  CVTT(rpeT, rpe_wf, DE, SEG, DE, SEG); CVTT(srpeT, srpe_wf, DE, SEG, DE, SEG);
  for (int m = 0; m < 2; ++m){
    CVTT(ipT[m], minp[m], DE, 2*DI, DE, 2*DI);
    CVTT(xpT[m], mxp[m], DI, PVHN, DI, 48);
    CVTT(opT[m], mop[m], DI, DE, DI, DE);
  }
  CVTT(w1T[0], W1_0, DG, DG, DG, DG); CVTT(w2T[0], W2_0, DG, DG, DG, DG);
  CVTT(w1T[1], W1_1, DG, DG, DG, DG); CVTT(w2T[1], W2_1, DG, DG, DG, DG);
  CVTT(predT[0], predw[0], LL0,   PP, LL0,   PP);
  CVTT(predT[1], predw[1], LL0/2, PP, LL0/2, PP);
  CVTT(predT[2], predw[2], LL0/4, PP, LL0/4, PP);
  k_prep_eps<<<1, 32, 0, stream>>>(eps0, eps1, eps1p);

  // ---- decomposition + trend head ----
  long totx = (long)BB * LL0 * CC;
  k_decomp<<<gsb(totx), 256, 0, stream>>>(x, res, trend, totx);
  k_trendT<<<gsb(totx), 256, 0, stream>>>(trend, trendTb, totx);
  GEMM(stream, trendTb, LL0, twT, LL0, trendo, nullptr, PP,
       (long)BB * CC, PP, LL0, trend_b, nullptr, nullptr, nullptr, 0, 0);
  k_down<<<gsb(totx/2), 256, 0, stream>>>(res, xs1, LL0/2, totx/2);
  k_down<<<gsb(totx/4), 256, 0, stream>>>(xs1, xs2, LL0/4, totx/4);

  const float* xsArr[3] = {res, xs1, xs2};
  const int Ls[3] = {512, 256, 128};
  const int Ns[3] = {1024, 512, 256};

  for (int i = 0; i < 3; ++i){
    const int L = Ls[i], N = Ns[i], Lp = L / SEG;
    const long Mr   = (long)BB * CC * Lp;   // mamba rows (mult of 64)
    const long Mgin = (long)N * BB * SEG;   // GIN MLP rows (mult of 64)
    const int ms = (i == 0) ? 0 : 1;
    const float* peb  = (i == 0) ? pe_b : spe_b;
    unsigned short* pewT  = (i == 0) ? peT : speT;
    unsigned short* rpewT = (i == 0) ? rpeT : srpeT;
    const float* rpebv = (i == 0) ? rpe_b : srpe_b;

    // ---- adaptive adjacency ----
    k_n12<<<cdivu((long)N * 64, 256), 256, 0, stream>>>(e1s[i], l1s[i], n1b, N);
    k_n12<<<cdivu((long)N * 64, 256), 256, 0, stream>>>(e2s[i], l2s[i], n2b, N);
    GEMM(stream, n1b, 64, n2b, 64, adjC, nullptr, N,
         N, N, 64, nullptr, nullptr, nullptr, nullptr, 0, 0);     // C1 = n1 n2^T
    k_adj<<<gsb((long)N * N), 256, 0, stream>>>(adjC, adjM, adjW, N);
    k_topk<<<N, 256, 0, stream>>>(adjM, adjW, adjB, N);

    // ---- GIN (2 layers), buffers rotate G0/G1/G2 ----
    long tH = (long)N * COLS;
    k_h0<<<gsb(tH), 256, 0, stream>>>(xsArr[i], start_w, start_b, G0, G1, L, N, tH);
    GEMM(stream, adjB, N, G1, N, nullptr, G2, COLS,
         N, COLS, N, nullptr, G0, eps1p + 0, nullptr, 0, 0);            // z1 -> G2
    GEMM(stream, G2, DG, w1T[0], DG, nullptr, G1, DG,
         Mgin, DG, DG, b1_0, nullptr, nullptr, nullptr, 1, 0);          // t1 -> G1
    GEMM(stream, G1, DG, w2T[0], DG, nullptr, G2, DG,
         Mgin, DG, DG, b2_0, nullptr, nullptr, nullptr, 1, 0);          // h1 -> G2
    k_transposeH<<<gsb(tH), 256, 0, stream>>>(G2, G1, N, tH);           // HT1 -> G1
    GEMM(stream, adjB, N, G1, N, nullptr, G0, COLS,
         N, COLS, N, nullptr, G2, eps1p + 1, nullptr, 0, 0);            // z2 -> G0
    GEMM(stream, G0, DG, w1T[1], DG, nullptr, G1, DG,
         Mgin, DG, DG, b1_1, nullptr, nullptr, nullptr, 1, 0);          // t2 -> G1
    GEMM(stream, G1, DG, w2T[1], DG, nullptr, G0, DG,
         Mgin, DG, DG, b2_1, nullptr, nullptr, nullptr, 1, 0);          // h2 -> G0
    long tXR = (long)BB * CC * L;
    k_zero_us<<<gsb(Mr * 32), 256, 0, stream>>>(xrb, Mr * 32);
    k_xr<<<gsb(tXR), 256, 0, stream>>>(G0, xsArr[i], start_w, start_b,
                                       re_w, re_b, xrb, L, N, Lp, tXR);

    // ---- Mamba block ----
    GEMM(stream, xrb, 32, pewT, 32, ubuf, nullptr, DE,
         Mr, DE, 32, peb, nullptr, nullptr, nullptr, 0, 0);             // emb = u
    k_rmsnorm<<<(unsigned)Mr, 64, 0, stream>>>(ubuf, mnorm[ms], xnb);
    GEMM(stream, xnb, DE, ipT[ms], DE, xzb, nullptr, 2 * DI,
         Mr, 2 * DI, DE, nullptr, nullptr, nullptr, nullptr, 0, 0);     // xz
    k_conv<<<gsb(Mr * DI), 256, 0, stream>>>(xzb, mcw[ms], mcb[ms], xcf, xcb, Lp, Mr * DI);
    GEMM(stream, xcb, DI, xpT[ms], DI, dbc, nullptr, 48,
         Mr, 48, DI, nullptr, nullptr, nullptr, nullptr, 0, 0);         // dbc (padded 48)
    k_dtk<<<gsb(Mr * DI), 256, 0, stream>>>(dbc, mdtw[ms], mdtb[ms], dtb, Mr * DI);
    k_scan<<<BB * CC, DI, 0, stream>>>(dtb, dbc, xcf, xzb, mAlog[ms], mD[ms], gbb, Lp);
    GEMM(stream, gbb, DI, opT[ms], DI, nullptr, encb, DE,
         Mr, DE, DI, nullptr, nullptr, nullptr, ubuf, 0, 0);            // enc = u + ...
    GEMM(stream, encb, DE, rpewT, DE, nullptr, ybf, SEG,
         Mr, SEG, DE, rpebv, nullptr, nullptr, nullptr, 0, 0);          // dec -> y
    GEMM(stream, ybf, L, predT[i], L, accp, nullptr, PP,
         (long)BB * CC, PP, L, predb[i], nullptr, nullptr, nullptr, 0, (i > 0) ? 1 : 0);
  }

  long tout = (long)BB * PP * CC;
  k_final<<<gsb(tout), 256, 0, stream>>>(accp, trendo, (float*)d_out, tout);
}
